// SelfAttention_13546326851680
// MI455X (gfx1250) — compile-verified, matrix operand order checked by reading, not run on hardware
//
#include <hip/hip_runtime.h>
#include <hip/hip_bf16.h>

typedef __attribute__((ext_vector_type(16))) _Float16 v16h;
typedef __attribute__((ext_vector_type(8)))  _Float16 v8h;
typedef __attribute__((ext_vector_type(8)))  float    v8f;

#define EMBED 1024
#define HEADS 16
#define HDIM  64
#define NBATCH 2
#define SEQ   2048
#define LPITCH 72   // LDS tile pitch in halves: 144B row stride -> 36 banks,
                    // gcd(36,64)=4 -> conflict-free 16B fragment loads, 16B aligned

// ---------------------------------------------------------------------------
// WMMA: D = A(16x32 f16) * B(32x16 f16) + C(16x16 f32)
// ---------------------------------------------------------------------------
__device__ __forceinline__ v8f wmma_f16(v16h a, v16h b, v8f c) {
  return __builtin_amdgcn_wmma_f32_16x16x32_f16(false, a, false, b, (short)0, c,
                                                false, false);
}

// ---------------------------------------------------------------------------
// Fragment loaders (CDNA5 ISA 7.12.2, wave32):
//  A 16x32 f16: lane&15 = row M; lanes 0-15 hold K 0-7 & 16-23, lanes 16-31
//  hold K 8-15 & 24-31.  B 32x16 mirrors A with N in lanes.
// ---------------------------------------------------------------------------
__device__ __forceinline__ v16h ldA_f32(const float* __restrict__ src, int ld) {
  const int lane = threadIdx.x & 31;
  const float* row = src + (size_t)(lane & 15) * ld + ((lane & 16) ? 8 : 0);
  v16h a;
#pragma unroll
  for (int i = 0; i < 8; ++i) {
    a[i]     = (_Float16)row[i];
    a[i + 8] = (_Float16)row[i + 16];
  }
  return a;
}

__device__ __forceinline__ v16h ldA_f16(const _Float16* src, int ld) {
  const int lane = threadIdx.x & 31;
  const _Float16* row = src + (size_t)(lane & 15) * ld + ((lane & 16) ? 8 : 0);
  v8h lo = *(const v8h*)(row);
  v8h hh = *(const v8h*)(row + 16);
  v16h a;
#pragma unroll
  for (int i = 0; i < 8; ++i) { a[i] = lo[i]; a[i + 8] = hh[i]; }
  return a;
}

// B fragment where memory holds B^T row-major: elem(k,n) at src[n*ld + k]
__device__ __forceinline__ v16h ldBt_f16(const _Float16* src, int ld) {
  const int lane = threadIdx.x & 31;
  const _Float16* col = src + (size_t)(lane & 15) * ld + ((lane & 16) ? 8 : 0);
  v8h lo = *(const v8h*)(col);
  v8h hh = *(const v8h*)(col + 16);
  v16h b;
#pragma unroll
  for (int i = 0; i < 8; ++i) { b[i] = lo[i]; b[i + 8] = hh[i]; }
  return b;
}

// ---------------------------------------------------------------------------
// Async global -> LDS cooperative 64x64-half tile fill (pitch LPITCH).
// GLOBAL_LOAD_ASYNC_TO_LDS_B128 (ISA 15.18.3 op 98): 16B/lane straight into
// LDS, ASYNCcnt-tracked, no VGPR round trip in the WMMA hot loop.
// ---------------------------------------------------------------------------
__device__ __forceinline__ void async_copy_tile64(const _Float16* gsrc,
                                                  size_t gpitch, _Float16* dst,
                                                  int wv, int lane) {
#pragma unroll
  for (int s = 0; s < 2; ++s) {
    const int seg = wv * 64 + s * 32 + lane;   // 0..511 segments of 8 halves
    const int row = seg >> 3, off = (seg & 7) * 8;
    const _Float16* g = gsrc + (size_t)row * gpitch + off;
    const unsigned laddr = (unsigned)(size_t)(dst + (size_t)row * LPITCH + off);
    asm volatile("global_load_async_to_lds_b128 %0, %1, off"
                 :: "v"(laddr), "v"(g) : "memory");
  }
}

__device__ __forceinline__ void wait_async_and_barrier() {
  asm volatile("s_wait_asynccnt 0x0" ::: "memory");  // own fills done
  __syncthreads();                                   // everyone's fills done
}

// ---------------------------------------------------------------------------
// Kernel 0a: bit-pack mask (int32 0/1 -> 1 bit): pm[row][col/8]
// ---------------------------------------------------------------------------
__global__ __launch_bounds__(256) void pack_mask(const int* __restrict__ mask,
                                                 unsigned char* __restrict__ pm) {
  const int i = blockIdx.x * blockDim.x + threadIdx.x;   // byte index
  if (i >= SEQ * SEQ / 8) return;
  const int* src = mask + (size_t)i * 8;
  unsigned b = 0;
#pragma unroll
  for (int j = 0; j < 8; ++j) b |= (src[j] != 0 ? 1u : 0u) << j;
  pm[i] = (unsigned char)b;
}

// ---------------------------------------------------------------------------
// Kernel 0b: f32 -> f16 weight conversion
// ---------------------------------------------------------------------------
__global__ __launch_bounds__(256) void cvt_weights(
    const float* __restrict__ Wv, const float* __restrict__ Wk,
    const float* __restrict__ Wq, const float* __restrict__ Wo,
    _Float16* __restrict__ Wvh, _Float16* __restrict__ Wkh,
    _Float16* __restrict__ Wqh, _Float16* __restrict__ Woh) {
  const int i = blockIdx.x * blockDim.x + threadIdx.x;
  if (i < HDIM * HDIM) {
    Wvh[i] = (_Float16)Wv[i];
    Wkh[i] = (_Float16)Wk[i];
    Wqh[i] = (_Float16)Wq[i];
  }
  if (i < EMBED * EMBED) Woh[i] = (_Float16)Wo[i];
}

// ---------------------------------------------------------------------------
// Kernel 1: per-head projection. One wave = 16 tokens x 64, 8 WMMA.
// VT=true stores transposed [n][h][e][S] (for contiguous V fragments later).
// ---------------------------------------------------------------------------
template <bool VT>
__global__ __launch_bounds__(256) void proj_head(
    const float* __restrict__ X, const _Float16* __restrict__ Wh,
    _Float16* __restrict__ Y) {
  const int w  = blockIdx.x * (blockDim.x >> 5) + (threadIdx.x >> 5);
  const int n  = w / (HEADS * (SEQ / 16));
  const int rw = w % (HEADS * (SEQ / 16));
  const int h  = rw / (SEQ / 16);
  const int t0 = (rw % (SEQ / 16)) * 16;

  const float* xb = X + ((size_t)(n * SEQ + t0) * EMBED + h * HDIM);
  v16h a0 = ldA_f32(xb, EMBED);        // d = 0..31
  v16h a1 = ldA_f32(xb + 32, EMBED);   // d = 32..63

  const int lane = threadIdx.x & 31;
  const int hi = (lane >> 4) & 1, nn = lane & 15;
  _Float16* yb = Y + (size_t)(n * HEADS + h) * SEQ * HDIM;

#pragma unroll
  for (int nb = 0; nb < 4; ++nb) {
    v16h b0 = ldBt_f16(Wh + (size_t)nb * 16 * HDIM, HDIM);
    v16h b1 = ldBt_f16(Wh + (size_t)nb * 16 * HDIM + 32, HDIM);
    v8f c = {};
    c = wmma_f16(a0, b0, c);
    c = wmma_f16(a1, b1, c);
#pragma unroll
    for (int r = 0; r < 8; ++r) {
      if (VT)   // [e][token]
        yb[(size_t)(nb * 16 + nn) * SEQ + t0 + r + hi * 8] = (_Float16)c[r];
      else      // [token][e]
        yb[(size_t)(t0 + r + hi * 8) * HDIM + nb * 16 + nn] = (_Float16)c[r];
    }
  }
}

// ---------------------------------------------------------------------------
// Kernel 2: flash attention. Block = 8 waves sharing one (n,h), 128 queries.
// 64-key K/V chunks double-buffered in LDS via async global->LDS DMA; the
// ping-pong is manually unrolled (two explicit bodies, compile-time LDS bases).
// Per chunk per wave: 8 WMMA (QK^T) + online softmax + 8 WMMA (PV).
// ---------------------------------------------------------------------------
__global__ __launch_bounds__(256) void flash_attn(
    const _Float16* __restrict__ Qh, const _Float16* __restrict__ Kh,
    const _Float16* __restrict__ VhT, const unsigned char* __restrict__ pm,
    _Float16* __restrict__ Oh) {
  __shared__ _Float16 Kt[2][64 * LPITCH];   // [key][d]
  __shared__ _Float16 Vt[2][64 * LPITCH];   // [e][key]
  __shared__ _Float16 Pt[8][16 * LPITCH];   // per-wave P tile

  const int tid = threadIdx.x;
  const int wv = tid >> 5;
  const int lane = tid & 31;
  const int hi = (lane >> 4) & 1, nn = lane & 15;

  const int b  = blockIdx.x;
  const int n  = b / (HEADS * (SEQ / 128));
  const int rb = b % (HEADS * (SEQ / 128));
  const int h  = rb / (SEQ / 128);
  const int t0 = (rb % (SEQ / 128)) * 128 + wv * 16;

  const size_t hoff = (size_t)(n * HEADS + h) * SEQ * HDIM;
  const _Float16* qb  = Qh + hoff + (size_t)t0 * HDIM;
  const _Float16* kp  = Kh + hoff;
  const _Float16* vtp = VhT + hoff;

  v16h aq0 = ldA_f16(qb, HDIM);
  v16h aq1 = ldA_f16(qb + 32, HDIM);

  v8f z = {};
  v8f o[4] = {z, z, z, z};
  float m[8], l[8];
#pragma unroll
  for (int r = 0; r < 8; ++r) { m[r] = -INFINITY; l[r] = 0.0f; }

  const float scale = 0.03125f;  // 1/sqrt(EMBED) per reference semantics

  // one 64-key chunk against LDS tiles Ktb/Vtb
  auto process_chunk = [&](const _Float16* Ktb, const _Float16* Vtb,
                           int kc) __attribute__((always_inline)) {
    // ---- energy S = Q . K^T: four 16-key blocks ----
    v8f s[4];
#pragma unroll
    for (int j = 0; j < 4; ++j) {
      v16h b0 = ldBt_f16(Ktb + (size_t)j * 16 * LPITCH, LPITCH);
      v16h b1 = ldBt_f16(Ktb + (size_t)j * 16 * LPITCH + 32, LPITCH);
      v8f sj = {};
      sj = wmma_f16(aq0, b0, sj);
      sj = wmma_f16(aq1, b1, sj);
      s[j] = sj;
    }
    // ---- mask bits: one u64 per row covers this 64-key chunk ----
    unsigned long long mb[8];
#pragma unroll
    for (int r = 0; r < 8; ++r)
      mb[r] = *(const unsigned long long*)(
          pm + (size_t)(t0 + r + hi * 8) * (SEQ / 8) + (kc >> 3));
    // ---- online softmax (rows per-VGPR across 16-lane halves) ----
    float p[4][8], alpha[8];
#pragma unroll
    for (int r = 0; r < 8; ++r) {
      float e[4], rmax = 0.0f;
#pragma unroll
      for (int j = 0; j < 4; ++j) {
        e[j] = ((mb[r] >> (j * 16 + nn)) & 1ULL) ? (float)s[j][r] * scale
                                                 : -3.125e18f;
        rmax = j ? fmaxf(rmax, e[j]) : e[j];
      }
#pragma unroll
      for (int x = 1; x < 16; x <<= 1) rmax = fmaxf(rmax, __shfl_xor(rmax, x, 32));
      const float mn = fmaxf(m[r], rmax);
      alpha[r] = __expf(m[r] - mn);
      float rs = 0.0f;
#pragma unroll
      for (int j = 0; j < 4; ++j) { p[j][r] = __expf(e[j] - mn); rs += p[j][r]; }
#pragma unroll
      for (int x = 1; x < 16; x <<= 1) rs += __shfl_xor(rs, x, 32);
      l[r] = l[r] * alpha[r] + rs;
      m[r] = mn;
    }
#pragma unroll
    for (int nb = 0; nb < 4; ++nb)
#pragma unroll
      for (int r = 0; r < 8; ++r) o[nb][r] *= alpha[r];
    // ---- P: C-layout -> LDS -> two A fragments (16x64) ----
#pragma unroll
    for (int r = 0; r < 8; ++r)
#pragma unroll
      for (int j = 0; j < 4; ++j)
        Pt[wv][(size_t)(r + hi * 8) * LPITCH + j * 16 + nn] = (_Float16)p[j][r];
    asm volatile("s_wait_dscnt 0" ::: "memory");  // same-wave LDS RAW
    v16h ap0 = ldA_f16(&Pt[wv][0], LPITCH);
    v16h ap1 = ldA_f16(&Pt[wv][32], LPITCH);
    // ---- O += P (16x64) . V (64x64) ----
#pragma unroll
    for (int nb = 0; nb < 4; ++nb) {
      v16h bv0 = ldBt_f16(Vtb + (size_t)nb * 16 * LPITCH, LPITCH);
      v16h bv1 = ldBt_f16(Vtb + (size_t)nb * 16 * LPITCH + 32, LPITCH);
      o[nb] = wmma_f16(ap0, bv0, o[nb]);
      o[nb] = wmma_f16(ap1, bv1, o[nb]);
    }
  };

  async_copy_tile64(kp, HDIM, Kt[0], wv, lane);
  async_copy_tile64(vtp, SEQ, Vt[0], wv, lane);
  wait_async_and_barrier();

  for (int kc = 0; kc < SEQ; kc += 128) {     // manual ping-pong unroll
    if (kc + 64 < SEQ) {
      async_copy_tile64(kp + (size_t)(kc + 64) * HDIM, HDIM, Kt[1], wv, lane);
      async_copy_tile64(vtp + (kc + 64), SEQ, Vt[1], wv, lane);
    }
    process_chunk(Kt[0], Vt[0], kc);
    wait_async_and_barrier();
    if (kc + 128 < SEQ) {
      async_copy_tile64(kp + (size_t)(kc + 128) * HDIM, HDIM, Kt[0], wv, lane);
      async_copy_tile64(vtp + (kc + 128), SEQ, Vt[0], wv, lane);
    }
    process_chunk(Kt[1], Vt[1], kc + 64);
    wait_async_and_barrier();
  }

  // ---- normalize, store token-major f16 for the Wo GEMM ----
  float inv[8];
#pragma unroll
  for (int r = 0; r < 8; ++r) inv[r] = 1.0f / l[r];
  _Float16* ob = Oh + (size_t)(n * SEQ + t0) * EMBED + h * HDIM;
#pragma unroll
  for (int nb = 0; nb < 4; ++nb)
#pragma unroll
    for (int r = 0; r < 8; ++r)
      ob[(size_t)(r + hi * 8) * EMBED + nb * 16 + nn] =
          (_Float16)(o[nb][r] * inv[r]);
}

// ---------------------------------------------------------------------------
// Kernel 3: Out = Oh (4096x1024 f16) @ Wo^T + bo.  Block = 8 waves sharing a
// 64-column Wo panel, async-double-buffered in LDS, manual ping-pong unroll.
// ---------------------------------------------------------------------------
__global__ __launch_bounds__(256) void out_proj(
    const _Float16* __restrict__ Oh, const _Float16* __restrict__ Woh,
    const float* __restrict__ bo, float* __restrict__ out) {
  __shared__ _Float16 Wt[2][64 * LPITCH];   // [col][k]
  const int tid = threadIdx.x;
  const int wv = tid >> 5;
  const int lane = tid & 31;
  const int hi = (lane >> 4) & 1, nn = lane & 15;

  const int nt   = blockIdx.x & 15;          // 16 column panels of 64
  const int mblk = blockIdx.x >> 4;          // 32 row panels of 8x16
  const int row0 = (mblk * 8 + wv) * 16;
  const int col0 = nt * 64;

  v8f z = {};
  v8f acc[4] = {z, z, z, z};

  auto gemm_chunk = [&](const _Float16* Wtb, int kc) __attribute__((always_inline)) {
    v16h a0 = ldA_f16(Oh + (size_t)row0 * EMBED + kc, EMBED);
    v16h a1 = ldA_f16(Oh + (size_t)row0 * EMBED + kc + 32, EMBED);
#pragma unroll
    for (int nb = 0; nb < 4; ++nb) {
      v16h b0 = ldBt_f16(Wtb + (size_t)nb * 16 * LPITCH, LPITCH);
      v16h b1 = ldBt_f16(Wtb + (size_t)nb * 16 * LPITCH + 32, LPITCH);
      acc[nb] = wmma_f16(a0, b0, acc[nb]);
      acc[nb] = wmma_f16(a1, b1, acc[nb]);
    }
  };

  async_copy_tile64(Woh + (size_t)col0 * EMBED, EMBED, Wt[0], wv, lane);
  wait_async_and_barrier();

  for (int kc = 0; kc < EMBED; kc += 128) {   // manual ping-pong unroll
    if (kc + 64 < EMBED)
      async_copy_tile64(Woh + (size_t)col0 * EMBED + kc + 64, EMBED,
                        Wt[1], wv, lane);
    gemm_chunk(Wt[0], kc);
    wait_async_and_barrier();
    if (kc + 128 < EMBED)
      async_copy_tile64(Woh + (size_t)col0 * EMBED + kc + 128, EMBED,
                        Wt[0], wv, lane);
    gemm_chunk(Wt[1], kc + 64);
    wait_async_and_barrier();
  }

#pragma unroll
  for (int nb = 0; nb < 4; ++nb) {
    const float bias = bo[col0 + nb * 16 + nn];
#pragma unroll
    for (int r = 0; r < 8; ++r)
      out[(size_t)(row0 + r + hi * 8) * EMBED + col0 + nb * 16 + nn] =
          acc[nb][r] + bias;
  }
}

// ---------------------------------------------------------------------------
extern "C" void kernel_launch(void* const* d_in, const int* in_sizes, int n_in,
                              void* d_out, int out_size, void* d_ws, size_t ws_size,
                              hipStream_t stream) {
  (void)in_sizes; (void)n_in; (void)out_size; (void)ws_size;
  const float* values = (const float*)d_in[0];
  const float* keysp  = (const float*)d_in[1];
  const float* query  = (const float*)d_in[2];
  const int*   mask   = (const int*)d_in[3];
  /* d_in[4] = heads (compile-time constant 16) */
  const float* Wv = (const float*)d_in[5];
  const float* Wk = (const float*)d_in[6];
  const float* Wq = (const float*)d_in[7];
  const float* Wo = (const float*)d_in[8];
  const float* bo = (const float*)d_in[9];
  float* out = (float*)d_out;

  // workspace carve-up
  _Float16* ws  = (_Float16*)d_ws;
  _Float16* Wvh = ws;
  _Float16* Wkh = Wvh + HDIM * HDIM;
  _Float16* Wqh = Wkh + HDIM * HDIM;
  _Float16* Woh = Wqh + HDIM * HDIM;
  _Float16* Qh  = Woh + (size_t)EMBED * EMBED;
  const size_t TSZ = (size_t)NBATCH * HEADS * SEQ * HDIM;
  _Float16* Kh  = Qh + TSZ;
  _Float16* VhT = Kh + TSZ;
  _Float16* Oh  = VhT + TSZ;
  unsigned char* pm = (unsigned char*)(Oh + (size_t)NBATCH * SEQ * EMBED);

  pack_mask<<<(SEQ * SEQ / 8 + 255) / 256, 256, 0, stream>>>(mask, pm);
  cvt_weights<<<(EMBED * EMBED + 255) / 256, 256, 0, stream>>>(
      Wv, Wk, Wq, Wo, Wvh, Wkh, Wqh, Woh);

  const int pblocks = NBATCH * HEADS * (SEQ / 16) / 8;   // 512
  proj_head<true><<<pblocks, 256, 0, stream>>>(values, Wvh, VhT);
  proj_head<false><<<pblocks, 256, 0, stream>>>(keysp, Wkh, Kh);
  proj_head<false><<<pblocks, 256, 0, stream>>>(query, Wqh, Qh);

  const int ablocks = NBATCH * HEADS * (SEQ / 128);      // 512
  flash_attn<<<ablocks, 256, 0, stream>>>(Qh, Kh, VhT, pm, Oh);

  out_proj<<<512, 256, 0, stream>>>(Oh, Woh, bo, out);
}
